// MSAPairWeightedAveraging_15496242004724
// MI455X (gfx1250) — matrix operand order repaired and checked by reading, not compiled, not run on hardware
//
#include <hip/hip_runtime.h>
#include <hip/hip_bf16.h>
#include <math.h>

typedef __attribute__((ext_vector_type(16))) _Float16 v16h;
typedef __attribute__((ext_vector_type(8)))  _Float16 v8h;
typedef __attribute__((ext_vector_type(4)))  _Float16 v4h;
typedef __attribute__((ext_vector_type(8)))  float    v8f;
typedef __attribute__((ext_vector_type(4)))  float    v4f;
typedef __attribute__((ext_vector_type(4)))  unsigned int v4u;

#define N_MSA 256
#define N_TOK 1024
#define CM    64
#define CC    32
#define CZ    128
#define HH    8
#define HC    256   // H*c
#define EPS   1e-5f

static __device__ __forceinline__ v8f wmma16(v16h a, v16h b, v8f c) {
  // D = A(16x32 f16) * B(32x16 f16) + C(16x16 f32)
  return __builtin_amdgcn_wmma_f32_16x16x32_f16(false, a, false, b, (short)0, c,
                                                false, false);
}

// Hardware approximate ops (single v_rcp_f32 / v_rsq_f32, ~1 ulp)
static __device__ __forceinline__ float fast_rcp(float x) {
  return __builtin_amdgcn_rcpf(x);
}
static __device__ __forceinline__ float fast_rsq(float x) {
  return __builtin_amdgcn_rsqf(x);
}
static __device__ __forceinline__ float fast_sigmoid(float x) {
  return fast_rcp(1.f + __expf(-x));
}

// Build A/B v16h fragment from two contiguous 8-half chunks.
static __device__ __forceinline__ v16h joinAB(v8h lo, v8h hi) {
  v16h f;
#pragma unroll
  for (int e = 0; e < 8; ++e) { f[e] = lo[e]; f[e + 8] = hi[e]; }
  return f;
}

// ---------------- weight f32 -> f16 convert ----------------
__global__ void __launch_bounds__(256) kConv(const float* __restrict__ a,
                                             _Float16* __restrict__ b, int n) {
  int i = blockIdx.x * 256 + threadIdx.x;
  if (i < n) b[i] = (_Float16)a[i];
}

// ---------------- kA: LN(m) + v/g projections (WMMA) ----------------
// v_t, g_t layout: [h][m][c][n] f16  (n = token index)
__global__ void __launch_bounds__(256) kA(const float* __restrict__ mIn,
    const float* __restrict__ lnw, const float* __restrict__ lnb,
    const _Float16* __restrict__ Wmv, const _Float16* __restrict__ Wmg,
    _Float16* __restrict__ v_t, _Float16* __restrict__ g_t)
{
  __shared__ _Float16 lds[256][CM];   // 32 KB normalized tokens
  const int tid = threadIdx.x;
  const long tokBase = (long)blockIdx.x * 256;

  // phase 1: layernorm one token per thread (two-pass, float4)
  {
    const float* row = mIn + (tokBase + tid) * CM;
    float s = 0.f, s2 = 0.f;
#pragma unroll
    for (int k = 0; k < CM; k += 4) {
      v4f x = *(const v4f*)(row + k);
#pragma unroll
      for (int e = 0; e < 4; ++e) { s += x[e]; s2 += x[e] * x[e]; }
    }
    const float mu  = s * (1.f / CM);
    const float inv = fast_rsq(s2 * (1.f / CM) - mu * mu + EPS);
#pragma unroll
    for (int k = 0; k < CM; k += 4) {
      v4f x = *(const v4f*)(row + k);
      v4h o;
#pragma unroll
      for (int e = 0; e < 4; ++e)
        o[e] = (_Float16)((x[e] - mu) * inv * lnw[k + e] + lnb[k + e]);
      *(v4h*)&lds[tid][k] = o;
    }
  }
  __syncthreads();

  // phase 2: [256 tok x 64] * [64 x 256]^T for W_mv and W_mg
  const int wave = tid >> 5, lane = tid & 31;
  const int lrow = lane & 15, lgrp = lane >> 4;
  const int mIdx  = (int)(tokBase >> 10);
  const int nBase = (int)(tokBase & 1023);
  for (int tile = wave; tile < 16; tile += 8) {
    const int t0 = tile * 16;
    v16h a0, a1;
    {
      v8h lo = *(const v8h*)&lds[t0 + lrow][16 * lgrp];
      v8h hi = *(const v8h*)&lds[t0 + lrow][16 * lgrp + 8];
      a0 = joinAB(lo, hi);
      lo = *(const v8h*)&lds[t0 + lrow][32 + 16 * lgrp];
      hi = *(const v8h*)&lds[t0 + lrow][32 + 16 * lgrp + 8];
      a1 = joinAB(lo, hi);
    }
    for (int n0 = 0; n0 < HC; n0 += 16) {
      const int outc = n0 + lrow;          // B col = output channel
      v8f accv = {}, accg = {};
      {
        v16h b = *(const v16h*)(Wmv + outc * CM + 16 * lgrp);
        accv = wmma16(a0, b, accv);
        b = *(const v16h*)(Wmv + outc * CM + 32 + 16 * lgrp);
        accv = wmma16(a1, b, accv);
      }
      {
        v16h b = *(const v16h*)(Wmg + outc * CM + 16 * lgrp);
        accg = wmma16(a0, b, accg);
        b = *(const v16h*)(Wmg + outc * CM + 32 + 16 * lgrp);
        accg = wmma16(a1, b, accg);
      }
      const int h = outc >> 5, c = outc & 31;
      const long base =
          (((long)(h * N_MSA + mIdx)) * CC + c) * N_TOK + nBase + t0 + 8 * lgrp;
      v8h pv, pg;
#pragma unroll
      for (int r = 0; r < 8; ++r) {
        pv[r] = (_Float16)accv[r];
        pg[r] = (_Float16)fast_sigmoid(accg[r]);   // sigmoid gate, v_rcp path
      }
      *(v8h*)(v_t + base) = pv;
      *(v8h*)(g_t + base) = pg;
    }
  }
}

// ---------------- kB: LN(z) + b = zn @ W_z^T  -> bbuf[h][i][j] ----------------
__global__ void __launch_bounds__(256) kB(const float* __restrict__ z,
    const float* __restrict__ lnw, const float* __restrict__ lnb,
    const float* __restrict__ Wz, float* __restrict__ bbuf)
{
  __shared__ float bp[8][8];   // [wave(j)][h]
  const int tid = threadIdx.x, wave = tid >> 5, lane = tid & 31;
  const long pair = (long)blockIdx.x * 8 + wave;    // i*1024 + j
  const float* row = z + pair * CZ;
  v4f x = *(const v4f*)(row + lane * 4);
  float s  = x[0] + x[1] + x[2] + x[3];
  float s2 = x[0]*x[0] + x[1]*x[1] + x[2]*x[2] + x[3]*x[3];
#pragma unroll
  for (int o = 16; o > 0; o >>= 1) {
    s  += __shfl_xor(s,  o, 32);
    s2 += __shfl_xor(s2, o, 32);
  }
  const float mu  = s * (1.f / CZ);
  const float inv = fast_rsq(s2 * (1.f / CZ) - mu * mu + EPS);
  float zn[4];
#pragma unroll
  for (int e = 0; e < 4; ++e) {
    const int k = lane * 4 + e;
    zn[e] = (x[e] - mu) * inv * lnw[k] + lnb[k];
  }
#pragma unroll
  for (int h = 0; h < HH; ++h) {
    const float* wr = Wz + h * CZ + lane * 4;
    float p = zn[0]*wr[0] + zn[1]*wr[1] + zn[2]*wr[2] + zn[3]*wr[3];
#pragma unroll
    for (int o = 16; o > 0; o >>= 1) p += __shfl_xor(p, o, 32);
    if (lane == 0) bp[wave][h] = p;
  }
  __syncthreads();
  if (tid < 64) {
    const int h = tid >> 3, jj = tid & 7;
    const long pr = (long)blockIdx.x * 8 + jj;
    const int ii = (int)(pr >> 10), jg = (int)(pr & 1023);
    bbuf[((long)h * N_TOK + ii) * N_TOK + jg] = bp[jj][h];
  }
}

// ---------------- kC: softmax over j -> wbuf[h][i][j] f16 ----------------
__global__ void __launch_bounds__(256) kC(const float* __restrict__ bbuf,
                                          _Float16* __restrict__ wbuf)
{
  __shared__ float red[8];
  const long rowbase = (long)blockIdx.x * N_TOK;   // blockIdx = h*1024 + i
  const int tid = threadIdx.x, wave = tid >> 5, lane = tid & 31;
  v4f x = *(const v4f*)(bbuf + rowbase + tid * 4);
  float mx = fmaxf(fmaxf(x[0], x[1]), fmaxf(x[2], x[3]));
#pragma unroll
  for (int o = 16; o > 0; o >>= 1) mx = fmaxf(mx, __shfl_xor(mx, o, 32));
  if (lane == 0) red[wave] = mx;
  __syncthreads();
  mx = red[0];
#pragma unroll
  for (int w = 1; w < 8; ++w) mx = fmaxf(mx, red[w]);
  float e0 = __expf(x[0]-mx), e1 = __expf(x[1]-mx);
  float e2 = __expf(x[2]-mx), e3 = __expf(x[3]-mx);
  float s = e0 + e1 + e2 + e3;
#pragma unroll
  for (int o = 16; o > 0; o >>= 1) s += __shfl_xor(s, o, 32);
  __syncthreads();
  if (lane == 0) red[wave] = s;
  __syncthreads();
  s = 0.f;
#pragma unroll
  for (int w = 0; w < 8; ++w) s += red[w];
  const float inv = fast_rcp(s);
  v4h o;
  o[0] = (_Float16)(e0*inv); o[1] = (_Float16)(e1*inv);
  o[2] = (_Float16)(e2*inv); o[3] = (_Float16)(e3*inv);
  *(v4h*)(wbuf + rowbase + tid * 4) = o;
}

// ---------------- kD: wv[m,i,h,c] = sum_j w[h,i,j]*v[h,m,c,j], gated ----------------
// writes wvg[token=(m,i)][h*32+c] f16
__global__ void __launch_bounds__(256) kD(const _Float16* __restrict__ wbuf,
    const _Float16* __restrict__ v_t, const _Float16* __restrict__ g_t,
    _Float16* __restrict__ wvg)
{
  __shared__ _Float16 lw[16][N_TOK];   // 32 KB: w tile, 16 i-rows x 1024 j
  const int bx = blockIdx.x;
  const int h  = bx >> 6;              // h-major: blocks sharing h reuse v_h in L2
  const int i0 = (bx & 63) * 16;
  const int tid = threadIdx.x;
  {
    const v4u* src = (const v4u*)(wbuf + ((long)h * N_TOK + i0) * N_TOK);
    v4u* dst = (v4u*)&lw[0][0];
#pragma unroll
    for (int idx = tid; idx < 16 * N_TOK / 8; idx += 256) dst[idx] = src[idx];
  }
  __syncthreads();

  const int wave = tid >> 5, lane = tid & 31, lrow = lane & 15, lgrp = lane >> 4;
  for (int m = wave; m < N_MSA; m += 8) {
    const _Float16* vbase = v_t + ((long)(h * N_MSA + m)) * CC * N_TOK;
    const _Float16* vcol0 = vbase + (long)lrow        * N_TOK + 16 * lgrp;
    const _Float16* vcol1 = vbase + (long)(16 + lrow) * N_TOK + 16 * lgrp;
    v8f acc0 = {}, acc1 = {};
    for (int j0 = 0; j0 < N_TOK; j0 += 32) {
      v8h lo = *(const v8h*)&lw[lrow][j0 + 16 * lgrp];
      v8h hi = *(const v8h*)&lw[lrow][j0 + 16 * lgrp + 8];
      v16h a  = joinAB(lo, hi);
      v16h b0 = *(const v16h*)(vcol0 + j0);
      v16h b1 = *(const v16h*)(vcol1 + j0);
      acc0 = wmma16(a, b0, acc0);
      acc1 = wmma16(a, b1, acc1);
    }
    if (m + 8 < N_MSA)   // CDNA5 global_prefetch of next m slice
      __builtin_prefetch(vcol0 + (long)8 * CC * N_TOK, 0, 1);

    // fused sigmoid gating + store
    const long gbase = ((long)(h * N_MSA + m)) * CC * N_TOK + i0 + 8 * lgrp;
    v8h g0 = *(const v8h*)(g_t + gbase + (long)lrow        * N_TOK);
    v8h g1 = *(const v8h*)(g_t + gbase + (long)(16 + lrow) * N_TOK);
    const long ob = ((long)m * N_TOK + i0 + 8 * lgrp) * HC + h * CC;
#pragma unroll
    for (int r = 0; r < 8; ++r) {
      wvg[ob + (long)r * HC + lrow]      = (_Float16)(acc0[r] * (float)g0[r]);
      wvg[ob + (long)r * HC + 16 + lrow] = (_Float16)(acc1[r] * (float)g1[r]);
    }
  }
}

// ---------------- kE: out = (g*wv) @ W_out^T  (K=256, N=64, f32 out) ----------------
__global__ void __launch_bounds__(256) kE(const _Float16* __restrict__ wvg,
    const _Float16* __restrict__ Wout, float* __restrict__ out)
{
  const int tid = threadIdx.x, wave = tid >> 5, lane = tid & 31;
  const int lrow = lane & 15, lgrp = lane >> 4;
  const long t0 = ((long)blockIdx.x * 8 + wave) * 16;   // 16-token tile
  const _Float16* arow = wvg + (t0 + lrow) * HC;
  v8f acc0 = {}, acc1 = {}, acc2 = {}, acc3 = {};
  for (int k0 = 0; k0 < HC; k0 += 32) {
    v8h lo = *(const v8h*)(arow + k0 + 16 * lgrp);
    v8h hi = *(const v8h*)(arow + k0 + 16 * lgrp + 8);
    v16h a  = joinAB(lo, hi);
    v16h b0 = *(const v16h*)(Wout + ( 0 + lrow) * HC + k0 + 16 * lgrp);
    v16h b1 = *(const v16h*)(Wout + (16 + lrow) * HC + k0 + 16 * lgrp);
    v16h b2 = *(const v16h*)(Wout + (32 + lrow) * HC + k0 + 16 * lgrp);
    v16h b3 = *(const v16h*)(Wout + (48 + lrow) * HC + k0 + 16 * lgrp);
    acc0 = wmma16(a, b0, acc0);
    acc1 = wmma16(a, b1, acc1);
    acc2 = wmma16(a, b2, acc2);
    acc3 = wmma16(a, b3, acc3);
  }
  const long ob = (t0 + 8 * lgrp) * CM + lrow;
#pragma unroll
  for (int r = 0; r < 8; ++r) {
    out[ob + (long)r * CM +  0] = acc0[r];
    out[ob + (long)r * CM + 16] = acc1[r];
    out[ob + (long)r * CM + 32] = acc2[r];
    out[ob + (long)r * CM + 48] = acc3[r];
  }
}

extern "C" void kernel_launch(void* const* d_in, const int* in_sizes, int n_in,
                              void* d_out, int out_size, void* d_ws, size_t ws_size,
                              hipStream_t stream) {
  (void)in_sizes; (void)n_in; (void)out_size; (void)ws_size;
  const float* m     = (const float*)d_in[0];
  const float* z     = (const float*)d_in[1];
  const float* ln_mw = (const float*)d_in[2];
  const float* ln_mb = (const float*)d_in[3];
  const float* ln_zw = (const float*)d_in[4];
  const float* ln_zb = (const float*)d_in[5];
  const float* W_mv  = (const float*)d_in[6];   // [256,64]
  const float* W_z   = (const float*)d_in[7];   // [8,128]
  const float* W_mg  = (const float*)d_in[8];   // [256,64]
  const float* W_out = (const float*)d_in[9];   // [64,256]
  float* out = (float*)d_out;

  char* ws = (char*)d_ws;
  _Float16* v_t    = (_Float16*)(ws);                          // 134217728 B
  _Float16* g_t    = (_Float16*)(ws + (size_t)134217728);      // 134217728 B
  _Float16* wvg    = (_Float16*)(ws + (size_t)268435456);      // 134217728 B
  float*    bbuf   = (float*)   (ws + (size_t)402653184);      //  33554432 B
  _Float16* wbuf   = (_Float16*)(ws + (size_t)436207616);      //  16777216 B
  _Float16* Wmv16  = (_Float16*)(ws + (size_t)452984832);
  _Float16* Wmg16  = (_Float16*)(ws + (size_t)452984832 + 32768);
  _Float16* Wout16 = (_Float16*)(ws + (size_t)452984832 + 65536);

  kConv<<<64, 256, 0, stream>>>(W_mv,  Wmv16,  16384);
  kConv<<<64, 256, 0, stream>>>(W_mg,  Wmg16,  16384);
  kConv<<<64, 256, 0, stream>>>(W_out, Wout16, 16384);
  kA<<<1024,   256, 0, stream>>>(m, ln_mw, ln_mb, Wmv16, Wmg16, v_t, g_t);
  kB<<<131072, 256, 0, stream>>>(z, ln_zw, ln_zb, W_z, bbuf);
  kC<<<8192,   256, 0, stream>>>(bbuf, wbuf);
  kD<<<512,    256, 0, stream>>>(wbuf, v_t, g_t, wvg);
  kE<<<2048,   256, 0, stream>>>(wvg, Wout16, out);
}
